// TransitionDown_78357383348684
// MI455X (gfx1250) — compile-verified, matrix operand order checked by reading, not run on hardware
//
#include <hip/hip_runtime.h>

// CDNA5 / gfx1250 wave32 WMMA implementation of TransitionDown.
// Pipeline (all on `stream`, serialized):
//   0) zero stat accumulators
//   1) KNN (LDS-tiled brute force, register top-16) + write new_pos output
//   2) GEMM pass 1 (async-gather->WMMA) accumulating per-channel sum/sumsq
//   3) BN param fold (scale/shift absorb linear bias)
//   4) GEMM pass 2: WMMA -> BN -> ReLU -> in-tile max over k (+ batch0 mean-over-k)
//   5) add broadcast batch0 mean term to all batches
// Per 16x16 tile: two v_wmma_f32_16x16x32_f16 (x-features, k=0..63) plus one
// v_wmma_f32_16x16x4_f32 (pos_enc, 3 live k) chained on the same accumulator.

typedef __attribute__((ext_vector_type(16))) _Float16 v16h;
typedef __attribute__((ext_vector_type(8)))  float    v8f;
typedef __attribute__((ext_vector_type(4)))  float    v4f;
typedef __attribute__((ext_vector_type(2)))  float    v2f;

#define B_     4
#define NFULL  16384
#define C_     64
#define NP     8192            // downsampled points per batch (8 frames * 1024)
#define OUT_   128
#define K_     16
#define MTILES (B_*NP)         // 32768 m-tiles; each = the 16 neighbors of one point
#define CNT    ((float)(B_*NP*K_))
#define EPS_   1e-5f

__device__ __forceinline__ int origIdx(int n) {
  // frame f = n>>10 maps to original frame 2f
  return ((n >> 10) << 11) | (n & 1023);
}

// wave-local sync after LDS stores: wave32 lockstep + own-DS-count wait.
#define WAVE_SYNC()                                            \
  do {                                                         \
    __builtin_amdgcn_wave_barrier();                           \
    asm volatile("s_wait_dscnt 0" ::: "memory");               \
    __builtin_amdgcn_wave_barrier();                           \
  } while (0)

// wave-local sync after async global->LDS staging (ASYNCcnt) + DS stores.
#define STAGE_SYNC()                                           \
  do {                                                         \
    __builtin_amdgcn_wave_barrier();                           \
    asm volatile("s_wait_asynccnt 0\n\ts_wait_dscnt 0" ::: "memory"); \
    __builtin_amdgcn_wave_barrier();                           \
  } while (0)

// CDNA5 async path: per-lane 16B global -> per-lane LDS, no VGPR data.
__device__ __forceinline__ void async_g2l_b128(unsigned lds_off, const void* gaddr) {
  asm volatile("global_load_async_to_lds_b128 %0, %1, off"
               :: "v"(lds_off), "v"((unsigned long long)(uintptr_t)gaddr)
               : "memory");
}

// ---------------------------------------------------------------- zero stats
__global__ void zero_kernel(float* g) {
  g[threadIdx.x] = 0.0f;   // 256 floats: gsum[128] + gsumq[128]
}

// ---------------------------------------------------------------- KNN
__global__ __launch_bounds__(256)
void knn_kernel(const float* __restrict__ pos, int* __restrict__ knn,
                float* __restrict__ newpos_out) {
  __shared__ float sp[2048 * 3];                 // 24 KB position tile
  const int batch = blockIdx.x >> 5;             // 32 blocks per batch
  const int q     = ((blockIdx.x & 31) << 8) + threadIdx.x;
  const float* pb = pos + (size_t)batch * NFULL * 3;
  const int oq    = origIdx(q);
  const float qx = pb[oq*3+0], qy = pb[oq*3+1], qz = pb[oq*3+2];

  // second tuple output: new_pos
  float* np_ = newpos_out + ((size_t)batch * NP + q) * 3;
  np_[0] = qx; np_[1] = qy; np_[2] = qz;

  float bd[K_]; int bi[K_];
#pragma unroll
  for (int i = 0; i < K_; ++i) { bd[i] = -3.4e38f; bi[i] = 0; }

  for (int c0 = 0; c0 < NP; c0 += 2048) {
    __syncthreads();
    for (int i = threadIdx.x; i < 2048; i += 256) {
      int op = origIdx(c0 + i);
      sp[i*3+0] = pb[op*3+0];
      sp[i*3+1] = pb[op*3+1];
      sp[i*3+2] = pb[op*3+2];
    }
    __syncthreads();
    for (int j = 0; j < 2048; ++j) {
      float dx = sp[j*3+0] - qx, dy = sp[j*3+1] - qy, dz = sp[j*3+2] - qz;
      float d = -(dx*dx + dy*dy + dz*dz);        // pd = -||pi-pj||^2 (maximize)
      if (d > bd[K_-1]) {
        bd[K_-1] = d; bi[K_-1] = c0 + j;
#pragma unroll
        for (int i = K_-1; i > 0; --i) {
          if (bd[i] > bd[i-1]) {                 // strict > keeps lower index on ties
            float t = bd[i]; bd[i] = bd[i-1]; bd[i-1] = t;
            int  ti = bi[i]; bi[i] = bi[i-1]; bi[i-1] = ti;
          }
        }
      }
    }
  }
  int* kr = knn + ((size_t)batch * NP + q) * K_;
#pragma unroll
  for (int i = 0; i < K_; ++i) kr[i] = bi[i];
}

// ---------------------------------------------------------------- GEMM passes
template <bool STATS>
__global__ __launch_bounds__(256)
void gemm_pass(const float* __restrict__ x, const float* __restrict__ pos,
               const float* __restrict__ W, const float* __restrict__ bias,
               const int* __restrict__ knn,
               float* __restrict__ gsum, float* __restrict__ gsumq,
               const float* __restrict__ scale, const float* __restrict__ shift,
               float* __restrict__ outmax, float* __restrict__ mean0) {
  // f16 B fragments pre-swizzled: [t(2)][j(8)][lane(32)][e(16)] halves = 16 KB
  __shared__ __align__(64) _Float16 Wswz[2 * 8 * 32 * 16];
  __shared__ __align__(16) float    Al[8][16 * 72];   // per-wave A slab (36 KB)
  __shared__ float sS[128], sQ[128];

  const int tid = threadIdx.x;
  // fill swizzled W for k=0..63 (all < 67, no guard needed)
  for (int i = tid; i < 2 * 8 * 32 * 16; i += 256) {
    int e = i & 15, ln = (i >> 4) & 31, j = (i >> 9) & 7, t = i >> 12;
    int c16 = ln & 15, h8 = (ln >> 4) << 3;
    int pair = e >> 1, off = e & 1;
    int kk = ((pair < 4) ? pair * 2 : 16 + (pair - 4) * 2) + h8 + off;
    Wswz[i] = (_Float16)W[(t * 32 + kk) * 128 + (j * 16 + c16)];
  }
  if (STATS && tid < 128) { sS[tid] = 0.0f; sQ[tid] = 0.0f; }
  __syncthreads();

  const int wv = tid >> 5, lane = tid & 31;
  const int col16 = lane & 15, hi = lane >> 4, hi8 = hi << 3;
  const int gw = blockIdx.x * 8 + wv;
  const int nwaves = 256 * 8;          // grid fixed at 256 WGs -> 16 tiles/wave exactly
  float* A = &Al[wv][0];
  const v16h* Bfrag = (const v16h*)Wswz;

  // hoist mt-invariant per-channel params + f32 WMMA B fragments (pos_enc rows).
  // 32-bit B 4x16 layout: lane = column, VGPR0 = K 2*hi, VGPR1 = K 2*hi+1.
  float pA[8], pB[8];
  v2f  b2[8];
  const int kr0 = 64 + 2 * hi, kr1 = 65 + 2 * hi;     // kr1==67 (hi) -> zero row
#pragma unroll
  for (int j = 0; j < 8; ++j) {
    int ch = j * 16 + col16;
    if (STATS) { pA[j] = bias[ch];  pB[j] = 0.0f; }
    else       { pA[j] = scale[ch]; pB[j] = shift[ch]; }
    b2[j][0] = W[kr0 * 128 + ch];                     // k = 64..66
    b2[j][1] = (kr1 < 67) ? W[kr1 * 128 + ch] : 0.0f; // pad k=67 with 0
  }

  for (int mt = gw; mt < MTILES; mt += nwaves) {
    const int b = mt >> 13, n = mt & (NP - 1);
    // neighbor indices live in lanes 0..15, broadcast via shfl
    int idxv = 0;
    if (lane < 16) idxv = knn[((size_t)b * NP + n) * K_ + lane];
    WAVE_SYNC();   // all LDS reads of previous tile complete before overwrite

    // async-stage grouped_x rows (16 x 64): per-lane global 16B -> LDS 16B
    for (int i = lane; i < 16 * 16; i += 32) {         // 256 float4s
      int r = i >> 4, c4 = i & 15;
      int nb = __shfl(idxv, r, 32);
      const float* src = &x[((size_t)b * NFULL + origIdx(nb)) * 64 + c4 * 4];
      async_g2l_b128((unsigned)(uintptr_t)&A[r * 72 + c4 * 4], src);
    }
    // pos_enc (cols 64..66) + zero pad (67..71)
    if (lane < 16) {
      int r = lane;
      int op = origIdx(idxv), oq = origIdx(n);
      const float* pb = pos + (size_t)b * NFULL * 3;
      A[r*72+64] = pb[op*3+0] - pb[oq*3+0];
      A[r*72+65] = pb[op*3+1] - pb[oq*3+1];
      A[r*72+66] = pb[op*3+2] - pb[oq*3+2];
      A[r*72+67] = 0.f; A[r*72+68] = 0.f; A[r*72+69] = 0.f;
      A[r*72+70] = 0.f; A[r*72+71] = 0.f;
    }
    STAGE_SYNC();

    // f16 A fragments: two runs of 8 consecutive k per lane
    // (k = t*32 + hi*8 + 0..7 and +16); 16B-aligned -> ds_load_b128.
    const float* Arow = A + col16 * 72;
    v16h afr[2];
#pragma unroll
    for (int t = 0; t < 2; ++t) {
      const v4f* p0 = (const v4f*)(Arow + t * 32 + hi8);
      const v4f* p1 = (const v4f*)(Arow + t * 32 + 16 + hi8);
      v4f a0 = p0[0], a1 = p0[1], b0 = p1[0], b1 = p1[1];
#pragma unroll
      for (int e = 0; e < 4; ++e) {
        afr[t][e]      = (_Float16)a0[e];
        afr[t][4 + e]  = (_Float16)a1[e];
        afr[t][8 + e]  = (_Float16)b0[e];
        afr[t][12 + e] = (_Float16)b1[e];
      }
    }
    // f32 A fragment for pos_enc: 32-bit A 16x4 layout, lane row = col16,
    // VGPR0 = K 2*hi, VGPR1 = K 2*hi+1 (k=67 reads LDS zero pad).
    v2f a2;
    a2[0] = Arow[64 + 2 * hi];
    a2[1] = Arow[65 + 2 * hi];

    // fully unrolled sweep of the 8 column tiles of OUT=128
#pragma unroll
    for (int j = 0; j < 8; ++j) {
      const int ch = j * 16 + col16;
      v8f c = {0.f, 0.f, 0.f, 0.f, 0.f, 0.f, 0.f, 0.f};
#pragma unroll
      for (int t = 0; t < 2; ++t) {
        v16h bfr = Bfrag[(t * 8 + j) * 32 + lane];   // one 32B contiguous LDS read
        c = __builtin_amdgcn_wmma_f32_16x16x32_f16(
                false, afr[t], false, bfr, (short)0, c, false, false);
      }
      // pos_enc contribution: K=4 f32 WMMA (3 live k; 4th padded zero)
      c = __builtin_amdgcn_wmma_f32_16x16x4_f32(
              false, a2, false, b2[j], (short)0, c, false, false);

      // C/D layout: lanes<16 hold rows(k) 0..7 in c[0..7], lanes>=16 rows 8..15;
      // column = ch. Reductions over k: per-lane 8-reg reduce + shfl_xor(16).
      if (STATS) {
        float bc = pA[j];
        float s = 0.f, qq = 0.f;
#pragma unroll
        for (int r = 0; r < 8; ++r) { float v = c[r] + bc; s += v; qq += v * v; }
        s  += __shfl_xor(s, 16, 32);
        qq += __shfl_xor(qq, 16, 32);
        if (hi == 0) { atomicAdd(&sS[ch], s); atomicAdd(&sQ[ch], qq); }
      } else {
        float sc = pA[j], sh = pB[j];            // linear bias folded into shift
        float m = -3.4e38f, s = 0.f;
#pragma unroll
        for (int r = 0; r < 8; ++r) {
          float v = fmaxf(c[r] * sc + sh, 0.0f); // BN + ReLU
          m = fmaxf(m, v); s += v;
        }
        m = fmaxf(m, __shfl_xor(m, 16, 32));
        s += __shfl_xor(s, 16, 32);
        if (hi == 0) {
          outmax[((size_t)(b * NP + n)) * OUT_ + ch] = m;
          if (b == 0) mean0[(size_t)n * OUT_ + ch] = s * (1.0f / 16.0f);
        }
      }
    }
  }

  if (STATS) {
    __syncthreads();
    if (tid < 128) { atomicAdd(&gsum[tid], sS[tid]); atomicAdd(&gsumq[tid], sQ[tid]); }
  }
}

// ---------------------------------------------------------------- BN fold
__global__ void bn_kernel(const float* __restrict__ gsum, const float* __restrict__ gsumq,
                          const float* __restrict__ bias, const float* __restrict__ gamma,
                          const float* __restrict__ beta,
                          float* __restrict__ scale, float* __restrict__ shift) {
  int c = threadIdx.x;
  float mean = gsum[c] / CNT;
  float var  = gsumq[c] / CNT - mean * mean;     // biased var, matches jnp.var
  float inv  = rsqrtf(var + EPS_);
  float sc   = gamma[c] * inv;
  scale[c] = sc;
  shift[c] = (bias[c] - mean) * sc + beta[c];    // h = d + bias -> d*sc + shift
}

// ---------------------------------------------------------------- mean broadcast add
__global__ __launch_bounds__(256)
void meanadd_kernel(float* __restrict__ out, const float* __restrict__ mean0) {
  const size_t total = (size_t)B_ * NP * OUT_;
  const size_t mask  = (size_t)NP * OUT_ - 1;    // NP*OUT_ is a power of two
  for (size_t i = (size_t)blockIdx.x * 256 + threadIdx.x; i < total;
       i += (size_t)gridDim.x * 256)
    out[i] += mean0[i & mask];
}

// ---------------------------------------------------------------- launcher
extern "C" void kernel_launch(void* const* d_in, const int* in_sizes, int n_in,
                              void* d_out, int out_size, void* d_ws, size_t ws_size,
                              hipStream_t stream) {
  const float* x     = (const float*)d_in[0];   // (4,16384,64)
  const float* pos   = (const float*)d_in[1];   // (4,16384,3)
  const float* W     = (const float*)d_in[2];   // (67,128)
  const float* bias  = (const float*)d_in[3];   // (128)
  const float* gamma = (const float*)d_in[4];   // (128)
  const float* beta  = (const float*)d_in[5];   // (128)

  float* out     = (float*)d_out;                         // (4,8192,128)
  float* outTail = out + (size_t)B_ * NP * OUT_;          // new_pos (4,8192,3)

  // workspace layout
  char*  ws    = (char*)d_ws;
  int*   knn   = (int*)ws;                                // 2 MB
  float* gsum  = (float*)(ws + 2097152);                  // 128
  float* gsumq = gsum + 128;
  float* scale = gsum + 256;
  float* shift = gsum + 384;
  float* mean0 = (float*)(ws + 2097152 + 2048);           // 4 MB

  zero_kernel<<<1, 256, 0, stream>>>(gsum);
  knn_kernel<<<128, 256, 0, stream>>>(pos, knn, outTail);
  gemm_pass<true><<<256, 256, 0, stream>>>(x, pos, W, bias, knn,
                                           gsum, gsumq, nullptr, nullptr,
                                           nullptr, nullptr);
  bn_kernel<<<1, 128, 0, stream>>>(gsum, gsumq, bias, gamma, beta, scale, shift);
  gemm_pass<false><<<256, 256, 0, stream>>>(x, pos, W, bias, knn,
                                            nullptr, nullptr, scale, shift,
                                            out, mean0);
  meanadd_kernel<<<2048, 256, 0, stream>>>(out, mean0);
}
